// Message_35502199669336
// MI455X (gfx1250) — compile-verified, matrix operand order checked by reading, not run on hardware
//
#include <hip/hip_runtime.h>
#include <hip/hip_bf16.h>

#define NFEAT   128
#define NNODES  10000
#define NEDGES  200000
#define TILE_E  32
#define NWAVES  8
#define PI_OVER_L 0.31415926535897931f

typedef __bf16 bf16;
typedef __attribute__((ext_vector_type(16))) __bf16 v16bf;
typedef __attribute__((ext_vector_type(8)))  __bf16 v8bf;
typedef __attribute__((ext_vector_type(8)))  float  v8f;

// Load a 16-element bf16 WMMA fragment for one lane.
// p points at element (row, kb + 8*h); elements 0..7 = K kb+8h..kb+8h+7,
// elements 8..15 = K kb+16+8h.. (per ISA 16-bit A/B VGPR layout, wave32).
__device__ __forceinline__ v16bf load_frag(const bf16* p) {
  v8bf lo = *(const v8bf*)p;
  v8bf hi = *(const v8bf*)(p + 16);
  v16bf r;
#pragma unroll
  for (int i = 0; i < 8; ++i) { r[i] = lo[i]; r[i + 8] = hi[i]; }
  return r;
}

__device__ __forceinline__ v8f wmma_bf16(v16bf a, v16bf b, v8f c) {
  return __builtin_amdgcn_wmma_f32_16x16x32_bf16(false, a, false, b, (short)0, c,
                                                 false, false);
}

__device__ __forceinline__ float silu(float x) {
  return x / (1.0f + __expf(-x));
}

// ---------------------------------------------------------------------------
// Kernel 1: fp32 weights -> bf16, transposed to N-major (Wt[n][k]) so WMMA
// B-fragments are two contiguous 16B loads per lane. Re-run every launch.
// ---------------------------------------------------------------------------
__global__ void convert_weights(const float* __restrict__ phiW1,  // (256,128)
                                const float* __restrict__ phiW2,  // (128,640)
                                const float* __restrict__ wW1,    // (128,128)
                                const float* __restrict__ wW2,    // (128,640)
                                bf16* __restrict__ ws) {
  bf16* pW1t = ws;                          // 128 x 256
  bf16* pW2t = ws + 32768;                  // 640 x 128
  bf16* qW1t = ws + 32768 + 81920;          // 128 x 128
  bf16* qW2t = ws + 32768 + 81920 + 16384;  // 640 x 128
  const int total = 32768 + 81920 + 16384 + 81920;
  for (int i = blockIdx.x * blockDim.x + threadIdx.x; i < total;
       i += gridDim.x * blockDim.x) {
    if (i < 32768) {
      int n = i >> 8, k = i & 255;
      pW1t[i] = (bf16)phiW1[k * 128 + n];
    } else if (i < 32768 + 81920) {
      int j = i - 32768;
      int n = j >> 7, k = j & 127;
      pW2t[j] = (bf16)phiW2[k * 640 + n];
    } else if (i < 32768 + 81920 + 16384) {
      int j = i - (32768 + 81920);
      int n = j >> 7, k = j & 127;
      qW1t[j] = (bf16)wW1[k * 128 + n];
    } else {
      int j = i - (32768 + 81920 + 16384);
      int n = j >> 7, k = j & 127;
      qW2t[j] = (bf16)wW2[k * 640 + n];
    }
  }
}

// ---------------------------------------------------------------------------
// Kernel 2: seed accumulated outputs (out_eq = eq, out_node = inv_node).
// ---------------------------------------------------------------------------
__global__ void init_out(const float* __restrict__ eq,       // N*F*3
                         const float* __restrict__ inv_node, // N*F
                         float* __restrict__ out) {
  const int total = NNODES * NFEAT * 3 + NNODES * NFEAT;
  for (int i = blockIdx.x * blockDim.x + threadIdx.x; i < total;
       i += gridDim.x * blockDim.x) {
    if (i < NNODES * NFEAT * 3) out[i] = eq[i];
    else out[i] = inv_node[i - NNODES * NFEAT * 3];
  }
}

// ---------------------------------------------------------------------------
// Kernel 3: fused edge pipeline. One block = 32 edges, 256 threads (8 wave32).
// Each wave owns N-tiles; B-fragments live in VGPRs and are reused across
// both 16-edge M-tiles, halving L2 weight traffic per WMMA.
// ---------------------------------------------------------------------------
__launch_bounds__(256)
__global__ void edge_kernel(const int* __restrict__ edge_index,   // (2,E)
                            const float* __restrict__ inv_node,   // (N,F)
                            const float* __restrict__ eq_node,    // (N,F,3)
                            const float* __restrict__ inv_edge,   // (E,F)
                            const float* __restrict__ dist,       // (E,)
                            const float* __restrict__ edir,       // (E,3)
                            const float* __restrict__ phi_b1,
                            const float* __restrict__ phi_b2,
                            const float* __restrict__ w_b1,
                            const float* __restrict__ w_b2,
                            const bf16* __restrict__ phiW1t,      // 128x256
                            const bf16* __restrict__ phiW2t,      // 640x128
                            const bf16* __restrict__ wW1t,        // 128x128
                            const bf16* __restrict__ wW2t,        // 640x128
                            float* __restrict__ eq_out,           // N*F*3
                            float* __restrict__ node_out,         // N*F
                            float* __restrict__ edge_out) {       // E*F
  __shared__ bf16  s_in[TILE_E * 256];   // concat(inv_node[src], inv_edge)
  __shared__ bf16  s_pe[TILE_E * 128];   // positional encoding
  __shared__ bf16  s_hp[TILE_E * 128];   // silu hidden (phi)
  __shared__ bf16  s_hw[TILE_E * 128];   // silu hidden (w)
  __shared__ float s_m3[TILE_E * 384];   // gates | cp_gates | scale
  __shared__ int   s_src[TILE_E], s_dst[TILE_E];
  __shared__ float s_dist[TILE_E], s_dir[TILE_E * 3];

  const int tid  = threadIdx.x;
  const int lane = tid & 31;
  const int wave = tid >> 5;
  const int e0   = blockIdx.x * TILE_E;

  // ---- phase 0: edge metadata ----
  if (tid < TILE_E) {
    s_src[tid]  = edge_index[e0 + tid];
    s_dst[tid]  = edge_index[NEDGES + e0 + tid];
    s_dist[tid] = dist[e0 + tid];
  }
  if (tid < TILE_E * 3) s_dir[tid] = edir[e0 * 3 + tid];
  __syncthreads();

  // ---- phase 1: build A matrices in LDS (bf16), float4-vectorized gather ----
  for (int i = 0; i < 8; ++i) {               // 32 x 256 in_feat, 4 elems/thread
    int q = i * 256 + tid;                    // vec4 index
    int e = q >> 6, k4 = (q & 63) * 4;        // k4 in [0,256), multiple of 4
    const float4 v = (k4 < NFEAT)
        ? *(const float4*)&inv_node[s_src[e] * NFEAT + k4]
        : *(const float4*)&inv_edge[(e0 + e) * NFEAT + (k4 - NFEAT)];
    bf16* d = &s_in[e * 256 + k4];
    d[0] = (bf16)v.x; d[1] = (bf16)v.y; d[2] = (bf16)v.z; d[3] = (bf16)v.w;
  }
  for (int i = 0; i < 16; ++i) {              // 32 x 128 pos-enc
    int idx = i * 256 + tid;
    int e = idx >> 7, j = idx & 127;
    float arg = s_dist[e] * (float)(j & 63) * PI_OVER_L;
    s_pe[idx] = (bf16)((j < 64) ? __sinf(arg) : __cosf(arg));
  }
  __syncthreads();

  const int m = lane & 15;     // row (A) or column (B/D)
  const int h = lane >> 4;     // half-wave selector

  // ---- phase 2: hidden layers. One N-tile per wave; B held in VGPRs ----
  {
    const int nt = wave;                      // 0..7
    const int n  = nt * 16 + m;
    // phi: h = silu(in_feat @ W1 + b1), K = 256
    v16bf bfr[8];
#pragma unroll
    for (int kb = 0; kb < 8; ++kb)
      bfr[kb] = load_frag(&phiW1t[n * 256 + kb * 32 + 8 * h]);
    float bias = phi_b1[n];
#pragma unroll
    for (int mt = 0; mt < 2; ++mt) {
      v8f acc = {};
#pragma unroll
      for (int kb = 0; kb < 8; ++kb) {
        v16bf a = load_frag(&s_in[(mt * 16 + m) * 256 + kb * 32 + 8 * h]);
        acc = wmma_bf16(a, bfr[kb], acc);
      }
#pragma unroll
      for (int r = 0; r < 8; ++r)
        s_hp[(mt * 16 + r + 8 * h) * NFEAT + n] = (bf16)silu(acc[r] + bias);
    }
    // w: hw = silu(pe @ W1 + b1), K = 128
    v16bf bfw[4];
#pragma unroll
    for (int kb = 0; kb < 4; ++kb)
      bfw[kb] = load_frag(&wW1t[n * 128 + kb * 32 + 8 * h]);
    bias = w_b1[n];
#pragma unroll
    for (int mt = 0; mt < 2; ++mt) {
      v8f acc = {};
#pragma unroll
      for (int kb = 0; kb < 4; ++kb) {
        v16bf a = load_frag(&s_pe[(mt * 16 + m) * 128 + kb * 32 + 8 * h]);
        acc = wmma_bf16(a, bfw[kb], acc);
      }
#pragma unroll
      for (int r = 0; r < 8; ++r)
        s_hw[(mt * 16 + r + 8 * h) * NFEAT + n] = (bf16)silu(acc[r] + bias);
    }
  }
  __syncthreads();

  // ---- phase 3: 640-wide output layers; fuse phi*w; dispatch by chunk ----
  for (int nt = wave; nt < 40; nt += NWAVES) {
    const int col = nt * 16 + m;
    v16bf bp[4], bw[4];
#pragma unroll
    for (int kb = 0; kb < 4; ++kb) {
      bp[kb] = load_frag(&phiW2t[col * NFEAT + kb * 32 + 8 * h]);
      bw[kb] = load_frag(&wW2t[col * NFEAT + kb * 32 + 8 * h]);
    }
    if (nt + NWAVES < 40) {   // hide L2 latency of the next weight tile
      __builtin_prefetch(&phiW2t[(col + NWAVES * 16) * NFEAT + 8 * h], 0, 1);
      __builtin_prefetch(&wW2t[(col + NWAVES * 16) * NFEAT + 8 * h], 0, 1);
    }
    const float bP = phi_b2[col], bW = w_b2[col];
    const int chunk = col >> 7;      // uniform per tile (128 % 16 == 0)
    const int f = col & 127;
#pragma unroll
    for (int mt = 0; mt < 2; ++mt) {
      v8f accP = {}, accW = {};
#pragma unroll
      for (int kb = 0; kb < 4; ++kb) {
        v16bf ap = load_frag(&s_hp[(mt * 16 + m) * NFEAT + kb * 32 + 8 * h]);
        accP = wmma_bf16(ap, bp[kb], accP);
        v16bf aw = load_frag(&s_hw[(mt * 16 + m) * NFEAT + kb * 32 + 8 * h]);
        accW = wmma_bf16(aw, bw[kb], accW);
      }
#pragma unroll
      for (int r = 0; r < 8; ++r) {
        int e = mt * 16 + r + 8 * h;
        float mv = (accP[r] + bP) * (accW[r] + bW);
        if (chunk < 3) {
          s_m3[e * 384 + chunk * NFEAT + f] = mv;         // gates/cp/scale
        } else if (chunk == 3) {
          atomicAdd(&node_out[s_dst[e] * NFEAT + f], mv); // ds segment-sum
        } else {
          int ge = e0 + e;                                // de: direct write
          edge_out[ge * NFEAT + f] = inv_edge[ge * NFEAT + f] + mv;
        }
      }
    }
  }
  __syncthreads();

  // ---- phase 4: equivariant update: scale*dir + gate*src_eq + cp*cross ----
  for (int i = 0; i < 16; ++i) {
    int idx = i * 256 + tid;
    int e = idx >> 7, f = idx & 127;
    float g  = s_m3[e * 384 + f];
    float cg = s_m3[e * 384 + 128 + f];
    float sc = s_m3[e * 384 + 256 + f];
    int sN = s_src[e], dN = s_dst[e];
    float ax = s_dir[e * 3 + 0], ay = s_dir[e * 3 + 1], az = s_dir[e * 3 + 2];
    const float* sp = &eq_node[(sN * NFEAT + f) * 3];
    const float* dp = &eq_node[(dN * NFEAT + f) * 3];
    float sx = sp[0], sy = sp[1], sz = sp[2];
    float dx = dp[0], dy = dp[1], dz = dp[2];
    float cx = ay * dz - az * dy;
    float cy = az * dx - ax * dz;
    float cz = ax * dy - ay * dx;
    float* op = &eq_out[(dN * NFEAT + f) * 3];
    atomicAdd(op + 0, sc * ax + g * sx + cg * cx);
    atomicAdd(op + 1, sc * ay + g * sy + cg * cy);
    atomicAdd(op + 2, sc * az + g * sz + cg * cz);
  }
}

// ---------------------------------------------------------------------------
extern "C" void kernel_launch(void* const* d_in, const int* in_sizes, int n_in,
                              void* d_out, int out_size, void* d_ws, size_t ws_size,
                              hipStream_t stream) {
  const int*   edge_index = (const int*)d_in[0];
  const float* inv_node   = (const float*)d_in[1];
  const float* eq_node    = (const float*)d_in[2];
  const float* inv_edge   = (const float*)d_in[3];
  const float* e_dist     = (const float*)d_in[4];
  const float* e_dir      = (const float*)d_in[5];
  const float* phi_W1     = (const float*)d_in[6];
  const float* phi_b1     = (const float*)d_in[7];
  const float* phi_W2     = (const float*)d_in[8];
  const float* phi_b2     = (const float*)d_in[9];
  const float* w_W1       = (const float*)d_in[10];
  const float* w_b1       = (const float*)d_in[11];
  const float* w_W2       = (const float*)d_in[12];
  const float* w_b2       = (const float*)d_in[13];

  float* out = (float*)d_out;
  float* eq_out   = out;                                        // N*F*3
  float* node_out = out + NNODES * NFEAT * 3;                   // N*F
  float* edge_out = out + NNODES * NFEAT * 3 + NNODES * NFEAT;  // E*F

  bf16* ws = (bf16*)d_ws;   // needs 425,984 bytes
  bf16* phiW1t = ws;
  bf16* phiW2t = ws + 32768;
  bf16* wW1t   = ws + 32768 + 81920;
  bf16* wW2t   = ws + 32768 + 81920 + 16384;

  convert_weights<<<208, 256, 0, stream>>>(phi_W1, phi_W2, w_W1, w_W2, ws);
  init_out<<<2560, 256, 0, stream>>>(eq_node, inv_node, out);
  edge_kernel<<<NEDGES / TILE_E, 256, 0, stream>>>(
      edge_index, inv_node, eq_node, inv_edge, e_dist, e_dir,
      phi_b1, phi_b2, w_b1, w_b2,
      phiW1t, phiW2t, wW1t, wW2t,
      eq_out, node_out, edge_out);
}
